// InteractionEmbedding_10565619548499
// MI455X (gfx1250) — matrix operand order repaired and checked by reading, not compiled, not run on hardware
//
#include <hip/hip_runtime.h>
#include <hip/hip_bf16.h>
#include <math.h>

// ---------------------------------------------------------------------------
// Fused upper-triangle-pair MLP for MI455X (gfx1250, wave32, WMMA f16->f32).
// rows = B*P = 256*8128 = 2,080,768 ; grid = rows/128 = 16,256 blocks of 256.
// Epilogue is fully branch-free: A&S erf (|err|<=1.5e-7) with v_rcp/v_exp.
// ---------------------------------------------------------------------------

typedef _Float16 v16h __attribute__((ext_vector_type(16)));
typedef _Float16 h8   __attribute__((ext_vector_type(8)));
typedef float    v8f  __attribute__((ext_vector_type(8)));

#define INVS 0.9995003748f /* 1/sqrt(1+0.001) */

// Branch-free exact-GELU: 0.5*z*(1+erf(z/sqrt(2))) with Abramowitz-Stegun 7.1.26.
// ~14 straight-line VALU ops (1 v_rcp_f32, 1 v_exp_f32), abs err ~1e-7.
__device__ __forceinline__ float gelu_fast(float z) {
    const float x  = z * 0.70710678118654752f;
    const float ax = __builtin_fabsf(x);
    const float t  = __builtin_amdgcn_rcpf(fmaf(0.3275911f, ax, 1.0f));
    float p = fmaf(t, 1.061405429f, -1.453152027f);
    p = fmaf(t, p, 1.421413741f);
    p = fmaf(t, p, -0.284496736f);
    p = fmaf(t, p, 0.254829592f);
    p = p * t;
    const float e  = __builtin_amdgcn_exp2f(ax * ax * -1.4426950408889634f);
    float er = fmaf(-p, e, 1.0f);            // erf(|x|)
    er = __builtin_copysignf(er, x);         // erf(x)
    const float hz = 0.5f * z;
    return fmaf(hz, er, hz);
}

// concatenate two aligned 8-half LDS segments into a 16-half WMMA fragment
__device__ __forceinline__ v16h load2x8h(const _Float16* p0, const _Float16* p1) {
    h8 lo = *(const h8*)p0;
    h8 hi = *(const h8*)p1;
    return __builtin_shufflevector(lo, hi, 0,1,2,3,4,5,6,7,8,9,10,11,12,13,14,15);
}

// One hidden layer: 16x64 (f16 in LDS staging) x 64x64 -> 16x64, gelu, back to staging.
// A fragment layout (16-bit A 16x32): lane L holds row M=L%16; halves 0..7 = K base+0..7,
// halves 8..15 = K base+16..23, base = 32*chunk + 8*(L/16).
// B fragment layout (16-bit B 32x16): lane L holds col N=L%16, K order same as A (base+0..7, base+16..23).
// P holds folded epilogue constants: P[n] = inv*g[n], P[64+n] = b[n]*inv*g[n] + be[n].
__device__ __forceinline__ void layer64(_Float16* stg, const _Float16* WT,
                                        const float* P, int half, int lm) {
    v16h A[2];
#pragma unroll
    for (int c = 0; c < 2; ++c) {
        const int base = lm * 72 + 32 * c + 8 * half;
        A[c] = load2x8h(&stg[base], &stg[base + 16]);
    }
#pragma unroll
    for (int t = 0; t < 4; ++t) {
        const int n = 16 * t + lm;
        v8f acc = {};
#pragma unroll
        for (int c = 0; c < 2; ++c) {
            const int o = n * 64 + 32 * c + 8 * half;
            v16h Bv = load2x8h(&WT[o], &WT[o + 16]);
            acc = __builtin_amdgcn_wmma_f32_16x16x32_f16(
                false, A[c], false, Bv, (short)0, acc, false, false);
        }
        const float aa = P[n], cc = P[64 + n];
#pragma unroll
        for (int r = 0; r < 8; ++r) {
            const float z = fmaf(acc[r], aa, cc);
            stg[(r + 8 * half) * 72 + n] = (_Float16)gelu_fast(z);
        }
    }
    // all lanes' staging stores must land before next layer's A-fragment reads
    asm volatile("s_wait_dscnt 0" ::: "memory");
}

__global__ void __launch_bounds__(256) fused_mlp(
    const float* __restrict__ inp,
    const float* __restrict__ W0, const float* __restrict__ b0,
    const float* __restrict__ W1, const float* __restrict__ b1,
    const float* __restrict__ W2, const float* __restrict__ b2,
    const float* __restrict__ W3, const float* __restrict__ b3,
    const float* __restrict__ g0, const float* __restrict__ e0,
    const float* __restrict__ g1, const float* __restrict__ e1,
    const float* __restrict__ g2, const float* __restrict__ e2,
    const float* __restrict__ g3, const float* __restrict__ e3,
    float* __restrict__ out)
{
    // weights transposed to [n][k] f16 so a B fragment is two contiguous 8-half runs
    __shared__ __attribute__((aligned(16))) _Float16 sWT0[64 * 16]; // K padded 4->16
    __shared__ __attribute__((aligned(16))) _Float16 sWT1[64 * 64];
    __shared__ __attribute__((aligned(16))) _Float16 sWT2[64 * 64];
    __shared__ __attribute__((aligned(16))) _Float16 sWT3[16 * 64]; // N padded 12->16
    __shared__ float sP0[2 * 64], sP1[2 * 64], sP2[2 * 64], sP3[2 * 16];
    __shared__ __attribute__((aligned(16))) _Float16 sStage[8 * 16 * 72]; // per-wave 16x64 (+pad)

    const int tid = threadIdx.x;

    for (int x = tid; x < 64 * 16; x += 256) sWT0[x] = (_Float16)0.0f;
    for (int x = tid; x < 16 * 64; x += 256) sWT3[x] = (_Float16)0.0f;
    __syncthreads();
    for (int x = tid; x < 4 * 64; x += 256) {
        const int k = x >> 6, n = x & 63;
        sWT0[n * 16 + k] = (_Float16)W0[x];
    }
    for (int x = tid; x < 64 * 64; x += 256) {
        const int k = x >> 6, n = x & 63;
        sWT1[n * 64 + k] = (_Float16)W1[x];
        sWT2[n * 64 + k] = (_Float16)W2[x];
    }
    for (int x = tid; x < 64 * 12; x += 256) {
        const int k = x / 12, n = x - k * 12;
        sWT3[n * 64 + k] = (_Float16)W3[x];
    }
    if (tid < 64) {
        // folded epilogue constants: z = acc*(inv*g) + (b*inv*g + be)
        const float a0 = INVS * g0[tid];
        sP0[tid] = a0; sP0[64 + tid] = fmaf(b0[tid], a0, e0[tid]);
        const float a1 = INVS * g1[tid];
        sP1[tid] = a1; sP1[64 + tid] = fmaf(b1[tid], a1, e1[tid]);
        const float a2 = INVS * g2[tid];
        sP2[tid] = a2; sP2[64 + tid] = fmaf(b2[tid], a2, e2[tid]);
    }
    if (tid < 16) {
        const float a3 = tid < 12 ? INVS * g3[tid] : 0.0f;
        sP3[tid]      = a3;
        sP3[16 + tid] = tid < 12 ? fmaf(b3[tid], a3, e3[tid]) : 0.0f;
    }
    __syncthreads();

    const int wave = tid >> 5;
    const int lane = tid & 31;
    const int half = lane >> 4;   // which 16-lane half of the wave
    const int lm   = lane & 15;
    _Float16* stg = sStage + wave * (16 * 72);

    // --- triangular index mapping for row m = lm (lanes 16..31 mirror 0..15) ---
    const int row  = (blockIdx.x * 8 + wave) * 16 + lm;     // < 2,080,768
    const int bidx = row / 8128;
    const int p    = row - bidx * 8128;                     // pair index, row-major triu(k=1)
    const float tt = sqrtf((float)(65025 - 8 * p));         // 255^2 - 8p
    int ii = (int)((255.0f - tt) * 0.5f);
    while ((((ii + 1) * (254 - ii)) >> 1) <= p) ++ii;       // start(i) = i*(255-i)/2
    while (((ii * (255 - ii)) >> 1) > p) --ii;
    const int jj = ii + 1 + (p - ((ii * (255 - ii)) >> 1));

    // --- layer 0: A = 16x32 f16 fragment, only K=0..3 nonzero (lanes 0..15) ---
    v16h A0 = {};
    if (half == 0) {
        const float4 x4 = *(const float4*)(inp + ((size_t)((bidx * 128 + ii) * 128 + jj) << 2));
        A0[0] = (_Float16)x4.x; A0[1] = (_Float16)x4.y;
        A0[2] = (_Float16)x4.z; A0[3] = (_Float16)x4.w;
    }
#pragma unroll
    for (int t = 0; t < 4; ++t) {
        const int n = 16 * t + lm;
        v16h Bv = {};                       // lanes 16..31 cover K=16..31 -> all zero
        if (half == 0) Bv = load2x8h(&sWT0[n * 16], &sWT0[n * 16 + 8]);
        v8f acc = {};
        acc = __builtin_amdgcn_wmma_f32_16x16x32_f16(
            false, A0, false, Bv, (short)0, acc, false, false);
        const float aa = sP0[n], cc = sP0[64 + n];
#pragma unroll
        for (int r = 0; r < 8; ++r) {
            const float z = fmaf(acc[r], aa, cc);
            stg[(r + 8 * half) * 72 + n] = (_Float16)gelu_fast(z);
        }
    }
    asm volatile("s_wait_dscnt 0" ::: "memory");

    // --- layers 1,2: 64 -> 64 ---
    layer64(stg, sWT1, sP1, half, lm);
    layer64(stg, sWT2, sP2, half, lm);

    // --- layer 3: 64 -> 12 (padded to 16), fused symmetric scatter ---
    v16h A[2];
#pragma unroll
    for (int c = 0; c < 2; ++c) {
        const int base = lm * 72 + 32 * c + 8 * half;
        A[c] = load2x8h(&stg[base], &stg[base + 16]);
    }
    v8f acc = {};
#pragma unroll
    for (int c = 0; c < 2; ++c) {
        const int o = lm * 64 + 32 * c + 8 * half;
        v16h Bv = load2x8h(&sWT3[o], &sWT3[o + 16]);
        acc = __builtin_amdgcn_wmma_f32_16x16x32_f16(
            false, A[c], false, Bv, (short)0, acc, false, false);
    }
    const int d = lm;
    const float aa = sP3[d], cc = sP3[16 + d];
#pragma unroll
    for (int r = 0; r < 8; ++r) {
        const int m  = r + 8 * half;                 // row within this wave's tile
        const int bm = __shfl(bidx, m, 32);
        const int im = __shfl(ii, m, 32);
        const int jm = __shfl(jj, m, 32);
        const float z = fmaf(acc[r], aa, cc);
        const float h = gelu_fast(z);
        if (d < 12) {
            out[((size_t)(bm * 128 + im) * 128 + jm) * 12 + d] = h;
            out[((size_t)(bm * 128 + jm) * 128 + im) * 12 + d] = h;
        }
    }
}

// out is poisoned by the harness; the diagonal (i==j) must be explicit zeros.
__global__ void __launch_bounds__(256) zero_diag(float* __restrict__ out) {
    const int idx = blockIdx.x * 256 + threadIdx.x;  // exactly 256*128*12 threads
    const int d = idx % 12;
    const int r = idx / 12;      // b*128 + i
    const int i = r & 127;
    const int b = r >> 7;
    out[((size_t)(b * 128 + i) * 128 + i) * 12 + d] = 0.0f;
}

extern "C" void kernel_launch(void* const* d_in, const int* in_sizes, int n_in,
                              void* d_out, int out_size, void* d_ws, size_t ws_size,
                              hipStream_t stream) {
    (void)in_sizes; (void)n_in; (void)d_ws; (void)ws_size; (void)out_size;
    const float* inp = (const float*)d_in[0];
    const float* W0  = (const float*)d_in[1];
    const float* b0  = (const float*)d_in[2];
    const float* W1  = (const float*)d_in[3];
    const float* b1  = (const float*)d_in[4];
    const float* W2  = (const float*)d_in[5];
    const float* b2  = (const float*)d_in[6];
    const float* W3  = (const float*)d_in[7];
    const float* b3  = (const float*)d_in[8];
    const float* g0  = (const float*)d_in[9];
    const float* e0  = (const float*)d_in[10];
    const float* g1  = (const float*)d_in[11];
    const float* e1  = (const float*)d_in[12];
    const float* g2  = (const float*)d_in[13];
    const float* e2  = (const float*)d_in[14];
    const float* g3  = (const float*)d_in[15];
    const float* e3  = (const float*)d_in[16];
    float* out = (float*)d_out;

    zero_diag<<<1536, 256, 0, stream>>>(out);                  // 393,216 diagonal elems
    fused_mlp<<<16256, 256, 0, stream>>>(inp, W0, b0, W1, b1, W2, b2, W3, b3,
                                         g0, e0, g1, e1, g2, e2, g3, e3, out);
}
